// GATC_12455405159095
// MI455X (gfx1250) — matrix operand order repaired and checked by reading, not compiled, not run on hardware
//
#include <hip/hip_runtime.h>
#include <cstdint>
#include <cstddef>

typedef __attribute__((ext_vector_type(16))) _Float16 v16h;
typedef __attribute__((ext_vector_type(8)))  float    v8f;
typedef int v4i_ __attribute__((vector_size(16)));   // int4 for async builtin

#define GLOBAL_AS __attribute__((address_space(1)))
#define LDS_AS    __attribute__((address_space(3)))

// Async global->LDS copy (CDNA5, tracked by ASYNCcnt). Guarded: falls back to
// synchronous wide loads if the builtin is not declared by this toolchain.
#if defined(__gfx1250__) && __has_builtin(__builtin_amdgcn_global_load_async_to_lds_b128)
#define USE_ASYNC_LDS 1
#else
#define USE_ASYNC_LDS 0
#endif

__device__ __forceinline__ void wait_async0(){
#if USE_ASYNC_LDS
#if __has_builtin(__builtin_amdgcn_s_wait_asynccnt)
  __builtin_amdgcn_s_wait_asynccnt(0);
#else
  asm volatile("s_wait_asynccnt 0x0" ::: "memory");
#endif
#endif
}

// ---------------- wave32 helpers (gfx1250 is wave32-only) ----------------
__device__ __forceinline__ float wave_max32(float v){
#pragma unroll
  for (int o = 16; o > 0; o >>= 1) v = fmaxf(v, __shfl_xor(v, o, 32));
  return v;
}
__device__ __forceinline__ float wave_sum32(float v){
#pragma unroll
  for (int o = 16; o > 0; o >>= 1) v += __shfl_xor(v, o, 32);
  return v;
}
__device__ __forceinline__ float lrelu(float x){ return x > 0.f ? x : 0.2f * x; }

// ---------------- f32 -> f16 cast ----------------
__global__ void k_cast_f16(const float* __restrict__ in, _Float16* __restrict__ out, int n){
  int i = blockIdx.x * blockDim.x + threadIdx.x;
  int stride = gridDim.x * blockDim.x;
  for (; i < n; i += stride) out[i] = (_Float16)in[i];
}

// ---------------- WMMA GEMM: C[M,N] = A[M,K](f16) @ B[K,N](f16), f32 out ----
// Block: 256 thr = 8 waves. Block tile 64(M) x 32(N), K-step 32, double-
// buffered LDS; A tile via async-to-LDS b128 (one barrier per K-step).
// Wave w: m_off=(w&3)*16, n_off=(w>>2)*16 -> one v_wmma_f32_16x16x32_f16 tile.
#define ASTRIDE 40   // halves; 80B rows: 16B aligned, 20-dword bank stride
#define BSTRIDE 36   // halves; 18-dword bank stride (transposed [n][k])

template<int K>
__global__ __launch_bounds__(256)
void k_gemm_wmma(const _Float16* __restrict__ A, const _Float16* __restrict__ B,
                 float* __restrict__ C, int M, int Nc){
  __shared__ _Float16 Ash[2][64 * ASTRIDE];
  __shared__ _Float16 Bsh[2][32 * BSTRIDE];

  const int tid  = threadIdx.x;
  const int lane = tid & 31;
  const int wv   = tid >> 5;
  const int m_off = (wv & 3) * 16;
  const int n_off = (wv >> 2) * 16;
  const int m0  = blockIdx.x * 64;
  const int nb0 = blockIdx.y * 32;

  // A tile: 64 rows x 32 halves = 4KB -> one 16B chunk per thread.
  // Row index clamped (rows >= M feed only unstored output rows) -> no branch.
  const int arow = tid >> 2;
  const int aqp  = tid & 3;
  int gr = m0 + arow; gr = (gr < M) ? gr : (M - 1);
  const _Float16* aSrcBase = A + (size_t)gr * K + aqp * 8;

  // B tile: 32(k) x 32(n) halves = 2KB -> one 8B chunk per thread, transposed
  const int bkr = tid >> 3;
  const int bnp = tid & 7;
  const _Float16* bSrcBase = B + (size_t)bkr * Nc + nb0 + bnp * 4;

  auto load_tiles = [&](int k0, int buf){
    // A: global b128 -> LDS b128 (async when available)
    const _Float16* asrc = aSrcBase + k0;
    _Float16* adst = &Ash[buf][arow * ASTRIDE + aqp * 8];
#if USE_ASYNC_LDS
    __builtin_amdgcn_global_load_async_to_lds_b128(
        (GLOBAL_AS v4i_*)asrc, (LDS_AS v4i_*)adst, 0, 0);
#else
    *(uint4*)adst = *(const uint4*)asrc;
#endif
    // B: global b64 -> 4 transposed half stores
    union { unsigned long long u; _Float16 h[4]; } cv;
    cv.u = *(const unsigned long long*)(bSrcBase + (size_t)k0 * Nc);
#pragma unroll
    for (int j = 0; j < 4; j++)
      Bsh[buf][(bnp * 4 + j) * BSTRIDE + bkr] = cv.h[j];
  };

  v8f acc = {};
  constexpr int NT = K / 32;
  load_tiles(0, 0);

#pragma unroll
  for (int t = 0; t < NT; ++t){
    const int buf = t & 1;
    wait_async0();           // this wave's async A copy for tile t landed
    __syncthreads();         // everyone's tile t visible; tile t-1 reads done
    if (t + 1 < NT){
      load_tiles((t + 1) * 32, buf ^ 1);
      __builtin_prefetch(bSrcBase + (size_t)(t + 1) * 32 * Nc, 0, 1);
    }
    // Fragments per ISA 16-bit A 16x32 / B 32x16 VGPR layouts.
    union { v16h v; unsigned int u[8]; } fa, fb;
    const int ml = m_off + (lane & 15);
    const int kg = (lane >> 4) * 8;           // lanes 16-31 hold K+8 group
#pragma unroll
    for (int j = 0; j < 4; j++)
      fa.u[j]     = *(const unsigned int*)&Ash[buf][ml * ASTRIDE + kg + 2 * j];
#pragma unroll
    for (int j = 0; j < 4; j++)
      fa.u[4 + j] = *(const unsigned int*)&Ash[buf][ml * ASTRIDE + 16 + kg + 2 * j];
    const int nl = n_off + (lane & 15);
    const int kb = (lane & 16) ? 8 : 0;       // lanes 16-31 hold K+8 group
#pragma unroll
    for (int j = 0; j < 8; j++)
      fb.u[j] = *(const unsigned int*)&Bsh[buf][nl * BSTRIDE + kb + 2 * j + ((j & 4) ? 8 : 0)];

    acc = __builtin_amdgcn_wmma_f32_16x16x32_f16(false, fa.v, false, fb.v,
                                                 (short)0, acc, false, false);
  }
  // D layout: VGPR r -> row = (lane>=16 ? 8 : 0)+r, col = lane&15
  const int col   = nb0 + n_off + (lane & 15);
  const int rbase = m0 + m_off + ((lane & 16) ? 8 : 0);
#pragma unroll
  for (int r = 0; r < 8; r++){
    int row = rbase + r;
    if (row < M) C[(size_t)row * Nc + col] = acc[r];
  }
}

// ---------------- degree + edge_attr segment-sum (for self-loop 'mean') ----
__global__ void k_deg_sumea(const int* __restrict__ dst, const float* __restrict__ ea,
                            int* __restrict__ deg, float* __restrict__ sum_ea, int E){
  int t = blockIdx.x * blockDim.x + threadIdx.x;
  if (t >= E * 16) return;
  int e = t >> 4, d = t & 15;
  int v = dst[e];
  atomicAdd(&sum_ea[v * 16 + d], ea[t]);
  if (d == 0) atomicAdd(&deg[v], 1);
}

// ---------------- hierarchical exclusive scan of deg -> row_start ----------
__global__ void k_scan1(const int* __restrict__ deg, int* __restrict__ incl,
                        int* __restrict__ bsums, int n){
  __shared__ int tmp[256];
  int base = blockIdx.x * 1024;
  int v[4]; int s = 0;
#pragma unroll
  for (int i = 0; i < 4; i++){
    int idx = base + threadIdx.x * 4 + i;
    v[i] = (idx < n) ? deg[idx] : 0;
    s += v[i];
  }
  tmp[threadIdx.x] = s;
  __syncthreads();
  for (int off = 1; off < 256; off <<= 1){
    int o = (threadIdx.x >= off) ? tmp[threadIdx.x - off] : 0;
    __syncthreads();
    tmp[threadIdx.x] += o;
    __syncthreads();
  }
  int run = tmp[threadIdx.x] - s;   // exclusive prefix of this thread
#pragma unroll
  for (int i = 0; i < 4; i++){
    int idx = base + threadIdx.x * 4 + i;
    run += v[i];
    if (idx < n) incl[idx] = run;   // inclusive per element
  }
  if (threadIdx.x == 255) bsums[blockIdx.x] = tmp[255];
}

__global__ void k_scan2(int* __restrict__ bsums, int nb){
  __shared__ int tmp[256];
  int v = (threadIdx.x < nb) ? bsums[threadIdx.x] : 0;
  tmp[threadIdx.x] = v;
  __syncthreads();
  for (int off = 1; off < 256; off <<= 1){
    int o = (threadIdx.x >= off) ? tmp[threadIdx.x - off] : 0;
    __syncthreads();
    tmp[threadIdx.x] += o;
    __syncthreads();
  }
  if (threadIdx.x < nb) bsums[threadIdx.x] = tmp[threadIdx.x];  // inclusive
}

__global__ void k_scan3(const int* __restrict__ incl, const int* __restrict__ bsums,
                        int* __restrict__ rs, int n){
  int idx = blockIdx.x * blockDim.x + threadIdx.x;
  if (idx >= n) return;
  int chunk = idx >> 10;
  int o = chunk ? bsums[chunk - 1] : 0;
  rs[idx + 1] = incl[idx] + o;
  if (idx == 0) rs[0] = 0;
}

// ---------------- CSR edge permutation by destination ----------------------
__global__ void k_perm(const int* __restrict__ dst, const int* __restrict__ rs,
                       int* __restrict__ cursor, int* __restrict__ perm, int E){
  int e = blockIdx.x * blockDim.x + threadIdx.x;
  if (e >= E) return;
  int v = dst[e];
  int pos = rs[v] + atomicAdd(&cursor[v], 1);
  perm[pos] = e;
}

// ---------------- fold We @ a_e into 16-float per head ---------------------
__global__ void k_we(const float* __restrict__ We, const float* __restrict__ a_e,
                     float* __restrict__ we, int Hn){
  int t = blockIdx.x * blockDim.x + threadIdx.x;
  if (t >= Hn * 16) return;
  int h = t >> 4, d = t & 15;
  int HC = Hn * 64;
  float s = 0.f;
  for (int c = 0; c < 64; c++) s += We[d * HC + h * 64 + c] * a_e[h * 64 + c];
  we[h * 16 + d] = s;
}

__global__ void k_aedge(const float* __restrict__ ea, const float* __restrict__ we,
                        float* __restrict__ aedge, int E, int Hn){
  int e = blockIdx.x * blockDim.x + threadIdx.x;
  if (e >= E) return;
  float r[16];
#pragma unroll
  for (int d = 0; d < 16; d++) r[d] = ea[e * 16 + d];
  for (int h = 0; h < Hn; h++){
    float s = 0.f;
#pragma unroll
    for (int d = 0; d < 16; d++) s += r[d] * we[h * 16 + d];
    aedge[e * Hn + h] = s;
  }
}

__global__ void k_aloop(const float* __restrict__ sum_ea, const int* __restrict__ deg,
                        const float* __restrict__ we, float* __restrict__ aloop,
                        int Nn, int Hn){
  int n = blockIdx.x * blockDim.x + threadIdx.x;
  if (n >= Nn) return;
  float inv = 1.f / fmaxf((float)deg[n], 1.f);
  float le[16];
#pragma unroll
  for (int d = 0; d < 16; d++) le[d] = sum_ea[n * 16 + d] * inv;
  for (int h = 0; h < Hn; h++){
    float s = 0.f;
#pragma unroll
    for (int d = 0; d < 16; d++) s += le[d] * we[h * 16 + d];
    aloop[n * Hn + h] = s;
  }
}

// ---------------- per-node attention source/dest scalars -------------------
__global__ void k_asad(const float* __restrict__ xp, const float* __restrict__ a_s,
                       const float* __restrict__ a_d, float* __restrict__ as_,
                       float* __restrict__ ad_, int Nn, int Hn){
  int t = blockIdx.x * blockDim.x + threadIdx.x;
  if (t >= Nn * Hn) return;
  int n = t / Hn, h = t % Hn;
  int HC = Hn * 64;
  const float* xr = xp + (size_t)n * HC + h * 64;
  float s1 = 0.f, s2 = 0.f;
  for (int c = 0; c < 64; c++){
    float v = xr[c];
    s1 += v * a_s[h * 64 + c];
    s2 += v * a_d[h * 64 + c];
  }
  as_[t] = s1; ad_[t] = s2;
}

// ---------------- wave-per-(node,head) segment softmax + aggregate ---------
// C=64 per head: lane owns channels {lane, lane+32}. No atomics (CSR order).
__global__ __launch_bounds__(256)
void k_agg(const float* __restrict__ xp, const int* __restrict__ srcv,
           const int* __restrict__ perm, const int* __restrict__ rs,
           const float* __restrict__ as_, const float* __restrict__ ad_,
           const float* __restrict__ aedge, const float* __restrict__ aloop,
           const float* __restrict__ bias, float* __restrict__ out,
           int Nn, int Hn){
  int wid = blockIdx.x * 8 + (threadIdx.x >> 5);
  if (wid >= Nn * Hn) return;
  int lane = threadIdx.x & 31;
  int n = wid / Hn, h = wid % Hn;
  int HC = Hn * 64;
  int row = rs[n], cnt = rs[n + 1] - row;
  float adn = ad_[n * Hn + h];
  float asn = as_[n * Hn + h];
  float alo = lrelu(asn + adn + aloop[n * Hn + h]);   // self-loop logit

  // phase 1: segment max
  float am = (lane == 0) ? alo : -3.4e38f;
  for (int i = lane; i < cnt; i += 32){
    int e = perm[row + i]; int s = srcv[e];
    am = fmaxf(am, lrelu(as_[s * Hn + h] + adn + aedge[e * Hn + h]));
  }
  am = wave_max32(am);

  // phase 2: denominator
  float den = (lane == 0) ? __expf(alo - am) : 0.f;
  for (int i = lane; i < cnt; i += 32){
    int e = perm[row + i]; int s = srcv[e];
    den += __expf(lrelu(as_[s * Hn + h] + adn + aedge[e * Hn + h]) - am);
  }
  den = wave_sum32(den);
  float inv = 1.f / (den + 1e-16f);

  // phase 3: weighted accumulate (xp fits in 192MB L2; gathers stay on-die)
  float acc0 = 0.f, acc1 = 0.f;
  for (int j = 0; j < cnt; j++){
    int e = perm[row + j]; int s = srcv[e];
    float wgt = __expf(lrelu(as_[s * Hn + h] + adn + aedge[e * Hn + h]) - am) * inv;
    const float* xr = xp + (size_t)s * HC + h * 64;
    acc0 += xr[lane]      * wgt;
    acc1 += xr[lane + 32] * wgt;
  }
  {
    float wgt = __expf(alo - am) * inv;
    const float* xr = xp + (size_t)n * HC + h * 64;
    acc0 += xr[lane] * wgt; acc1 += xr[lane + 32] * wgt;
  }
  float o0 = fmaxf(acc0 + bias[h * 64 + lane], 0.f);        // +b, ReLU
  float o1 = fmaxf(acc1 + bias[h * 64 + lane + 32], 0.f);
  out[(size_t)n * HC + h * 64 + lane]      = o0;
  out[(size_t)n * HC + h * 64 + lane + 32] = o1;
}

// ---------------- final linear + sigmoid -----------------------------------
__global__ void k_out(const float* __restrict__ h3, const float* __restrict__ lw,
                      const float* __restrict__ lb, float* __restrict__ out, int Nn){
  int n = blockIdx.x * blockDim.x + threadIdx.x;
  if (n >= Nn) return;
  float s = lb[0];
#pragma unroll 8
  for (int c = 0; c < 64; c++) s += h3[(size_t)n * 64 + c] * lw[c];
  out[n] = 1.f / (1.f + __expf(-s));
}

extern "C" void kernel_launch(void* const* d_in, const int* in_sizes, int n_in,
                              void* d_out, int out_size, void* d_ws, size_t ws_size,
                              hipStream_t stream){
  (void)n_in; (void)out_size; (void)ws_size;
  const float* x    = (const float*)d_in[0];
  const int*   ei   = (const int*)d_in[1];
  const float* ea   = (const float*)d_in[2];
  const float* W0   = (const float*)d_in[3];
  const float* as0  = (const float*)d_in[4];
  const float* ad0  = (const float*)d_in[5];
  const float* We0  = (const float*)d_in[6];
  const float* ae0  = (const float*)d_in[7];
  const float* b0   = (const float*)d_in[8];
  const float* W1   = (const float*)d_in[9];
  const float* as1  = (const float*)d_in[10];
  const float* ad1  = (const float*)d_in[11];
  const float* We1  = (const float*)d_in[12];
  const float* ae1  = (const float*)d_in[13];
  const float* b1   = (const float*)d_in[14];
  const float* W2   = (const float*)d_in[15];
  const float* as2  = (const float*)d_in[16];
  const float* ad2  = (const float*)d_in[17];
  const float* We2  = (const float*)d_in[18];
  const float* ae2  = (const float*)d_in[19];
  const float* b2   = (const float*)d_in[20];
  const float* linw = (const float*)d_in[21];
  const float* linb = (const float*)d_in[22];

  const int Nn = in_sizes[0] / 128;
  const int Ee = in_sizes[2] / 16;
  const int* srcv = ei;
  const int* dstv = ei + Ee;

  // bump allocator over d_ws
  char* base = (char*)d_ws; size_t off = 0;
  auto alloc = [&](size_t bytes) -> void* {
    off = (off + 255) & ~(size_t)255;
    void* p = base + off; off += bytes; return p;
  };
  float*    XP     = (float*)alloc((size_t)Nn * 256 * 4);
  float*    FA     = (float*)alloc((size_t)Nn * 256 * 4);
  float*    H3     = (float*)alloc((size_t)Nn * 64 * 4);
  _Float16* X16    = (_Float16*)alloc((size_t)Nn * 256 * 2);
  _Float16* W16    = (_Float16*)alloc((size_t)256 * 256 * 2);
  int*      deg    = (int*)alloc((size_t)Nn * 4);
  int*      rs     = (int*)alloc((size_t)(Nn + 1) * 4);
  int*      cursor = (int*)alloc((size_t)Nn * 4);
  int*      incl   = (int*)alloc((size_t)Nn * 4);
  int*      bsums  = (int*)alloc(256 * 4);
  int*      perm   = (int*)alloc((size_t)Ee * 4);
  float*    sum_ea = (float*)alloc((size_t)Nn * 16 * 4);
  float*    wEc    = (float*)alloc(4 * 16 * 4);
  float*    aedge  = (float*)alloc((size_t)Ee * 4 * 4);
  float*    aloop  = (float*)alloc((size_t)Nn * 4 * 4);
  float*    asb    = (float*)alloc((size_t)Nn * 4 * 4);
  float*    adb    = (float*)alloc((size_t)Nn * 4 * 4);

  (void)hipMemsetAsync(deg,    0, (size_t)Nn * 4,       stream);
  (void)hipMemsetAsync(cursor, 0, (size_t)Nn * 4,       stream);
  (void)hipMemsetAsync(sum_ea, 0, (size_t)Nn * 16 * 4,  stream);

  // graph structure (shared across layers)
  {
    int tot = Ee * 16;
    k_deg_sumea<<<(tot + 255) / 256, 256, 0, stream>>>(dstv, ea, deg, sum_ea, Ee);
    int nb = (Nn + 1023) / 1024;
    k_scan1<<<nb, 256, 0, stream>>>(deg, incl, bsums, Nn);
    k_scan2<<<1, 256, 0, stream>>>(bsums, nb);
    k_scan3<<<(Nn + 255) / 256, 256, 0, stream>>>(incl, bsums, rs, Nn);
    k_perm<<<(Ee + 255) / 256, 256, 0, stream>>>(dstv, rs, cursor, perm, Ee);
  }

  auto run_layer = [&](const float* feat_in, int K, int Hn,
                       const float* Wl, const float* a_sl, const float* a_dl,
                       const float* Wel, const float* a_el, const float* bl,
                       float* out_feat){
    int HC = Hn * 64;
    k_cast_f16<<<512, 256, 0, stream>>>(feat_in, X16, Nn * K);
    k_cast_f16<<<64, 256, 0, stream>>>(Wl, W16, K * HC);
    dim3 g((Nn + 63) / 64, HC / 32);
    if (K == 128)
      k_gemm_wmma<128><<<g, 256, 0, stream>>>(X16, W16, XP, Nn, HC);
    else
      k_gemm_wmma<256><<<g, 256, 0, stream>>>(X16, W16, XP, Nn, HC);
    k_we<<<1, 64, 0, stream>>>(Wel, a_el, wEc, Hn);
    k_aedge<<<(Ee + 255) / 256, 256, 0, stream>>>(ea, wEc, aedge, Ee, Hn);
    k_aloop<<<(Nn + 255) / 256, 256, 0, stream>>>(sum_ea, deg, wEc, aloop, Nn, Hn);
    k_asad<<<(Nn * Hn + 255) / 256, 256, 0, stream>>>(XP, a_sl, a_dl, asb, adb, Nn, Hn);
    int waves = Nn * Hn;
    k_agg<<<(waves + 7) / 8, 256, 0, stream>>>(XP, srcv, perm, rs, asb, adb,
                                               aedge, aloop, bl, out_feat, Nn, Hn);
  };

  run_layer(x,  128, 4, W0, as0, ad0, We0, ae0, b0, FA);
  run_layer(FA, 256, 4, W1, as1, ad1, We1, ae1, b1, FA);
  run_layer(FA, 256, 1, W2, as2, ad2, We2, ae2, b2, H3);

  k_out<<<(Nn + 255) / 256, 256, 0, stream>>>(H3, linw, linb, (float*)d_out, Nn);
}
